// SinkhornNet_20890720927839
// MI455X (gfx1250) — matrix operand order repaired and checked by reading, not compiled, not run on hardware
//
#include <hip/hip_runtime.h>
#include <math.h>

// ---------------------------------------------------------------------------
// SinkhornNet on MI455X (gfx1250, wave32, WMMA).
//
// All conv/linear layers run as implicit GEMM on v_wmma_f32_16x16x32_f16 with
// split-f16 (Markidis) operands pre-packed as uint32 {lo16(hi), hi16(residual)}:
// one b32 load per element yields both halves -> no conversion VALU in the
// K-loop. Weights for the 16-wide N-tile are staged into LDS with
// global_load_async_to_lds_* (ASYNCcnt) and shared by 4 waves per block.
// ---------------------------------------------------------------------------

typedef __attribute__((ext_vector_type(16))) _Float16 v16h;
typedef __attribute__((ext_vector_type(8)))  float    v8f;

static __device__ __forceinline__ unsigned pack_split(float x) {
    _Float16 h = (_Float16)x;
    _Float16 l = (_Float16)(x - (float)h);
    return ((unsigned)__builtin_bit_cast(unsigned short, l) << 16) |
           (unsigned)__builtin_bit_cast(unsigned short, h);
}
static __device__ __forceinline__ _Float16 lo_h(unsigned u) {
    return __builtin_bit_cast(_Float16, (unsigned short)(u & 0xffffu));
}
static __device__ __forceinline__ _Float16 hi_h(unsigned u) {
    return __builtin_bit_cast(_Float16, (unsigned short)(u >> 16));
}

__global__ void pack_split_kernel(const float* __restrict__ src,
                                  unsigned* __restrict__ dst, int n) {
    const int i = blockIdx.x * blockDim.x + threadIdx.x;
    if (i < n) dst[i] = pack_split(src[i]);
}

// Block = 128 threads = 4 waves. Each wave owns one 16(M)x16(N) WMMA tile;
// the 4 waves share one N-tile (16 output channels) whose weights are staged
// in LDS via async-to-LDS DMA. M = B*Hout*Wout, N = Cout, K = Cin*KH*KW.
template<int Cin, int Hin, int Win, int Cout, int Hout, int Wout,
         int KH, int KW, int STRIDE, bool RELU, bool PACK_OUT>
__global__ __launch_bounds__(128)
void conv_wmma_kernel(const unsigned* __restrict__ in,
                      const unsigned* __restrict__ wgt,
                      const float* __restrict__ bias,
                      void* __restrict__ outv)
{
    constexpr int KK    = KH * KW;
    constexpr int Ktot  = Cin * KK;
    constexpr int KPAD  = (Ktot + 31) & ~31;
    constexpr int KSLAB = (KPAD <= 512) ? KPAD : 512;   // LDS slab <= 32 KB
    constexpr int HW    = Hout * Wout;

    __shared__ unsigned ldsW[16 * KSLAB];

    const int tid   = threadIdx.x;
    const int wave  = tid >> 5;
    const int lane  = tid & 31;
    const int l16   = lane & 15;
    const int hsel  = lane >> 4;
    const int mtile = blockIdx.x * 4 + wave;
    const int ntile = blockIdx.y;

    // A-row carried by this lane (compile-time strength-reduced decode).
    const int m_a   = mtile * 16 + l16;
    const int n_img = m_a / HW;
    const int rem_a = m_a % HW;
    const int oh_a  = rem_a / Wout;
    const int ow_a  = rem_a % Wout;
    const int oc    = ntile * 16 + l16;

    const unsigned* wrow = wgt + ntile * 16 * Ktot;

    v8f acc = {};

    for (int ks = 0; ks < KPAD; ks += KSLAB) {
        __syncthreads();   // previous slab fully consumed
        if constexpr ((Ktot & 3) == 0) {
            // 16B-aligned rows: async B128 DMA into LDS.
            for (int idx = tid * 4; idx < 16 * KSLAB; idx += 512) {
                const int row = idx / KSLAB;
                const int col = ks + (idx % KSLAB);
                if (col < Ktot) {
                    const unsigned* src = wrow + row * Ktot + col;
                    const unsigned laddr = (unsigned)(unsigned long long)&ldsW[idx];
                    asm volatile("global_load_async_to_lds_b128 %0, %1, off"
                                 :: "v"(laddr), "v"(src) : "memory");
                }
            }
        } else {
            // K not a multiple of 4 (conv1, K=75): b32 DMA + zero-fill pad.
            for (int idx = tid; idx < 16 * KSLAB; idx += 128) {
                const int row = idx / KSLAB;
                const int col = ks + (idx % KSLAB);
                if (col < Ktot) {
                    const unsigned* src = wrow + row * Ktot + col;
                    const unsigned laddr = (unsigned)(unsigned long long)&ldsW[idx];
                    asm volatile("global_load_async_to_lds_b32 %0, %1, off"
                                 :: "v"(laddr), "v"(src) : "memory");
                } else {
                    ldsW[idx] = 0u;
                }
            }
        }
        asm volatile("s_wait_asynccnt 0x0" ::: "memory");
        __syncthreads();   // slab visible to all 4 waves

        const int kend = (ks + KSLAB < KPAD) ? (ks + KSLAB) : KPAD;
        for (int k0 = ks; k0 < kend; k0 += 32) {
            v16h a_hi = {}, a_lo = {}, b_hi = {}, b_lo = {};
            #pragma unroll
            for (int v = 0; v < 8; ++v) {
                const int kbA = 2 * v + 16 * hsel;
                const int kbB = 2 * v + 16 * hsel;
                #pragma unroll
                for (int e = 0; e < 2; ++e) {
                    // A element (packed split-f16) gathered im2col from global.
                    const int kA = k0 + kbA + e;
                    unsigned pa = 0u;
                    if (KPAD == Ktot || kA < Ktot) {
                        const int ic = kA / KK;
                        const int kr = kA % KK;
                        const int kh = kr / KW;
                        const int kw = kr % KW;
                        pa = in[((n_img * Cin + ic) * Hin + oh_a * STRIDE + kh) * Win
                                + ow_a * STRIDE + kw];
                    }
                    a_hi[2 * v + e] = lo_h(pa);
                    a_lo[2 * v + e] = hi_h(pa);
                    // B element from the LDS-staged weight slab.
                    const unsigned pb = ldsW[l16 * KSLAB + (k0 - ks) + kbB + e];
                    b_hi[2 * v + e] = lo_h(pb);
                    b_lo[2 * v + e] = hi_h(pb);
                }
            }
            // Split-f16 product: hi*hi + hi*lo + lo*hi (~22-bit mantissa).
            acc = __builtin_amdgcn_wmma_f32_16x16x32_f16(
                      false, a_hi, false, b_hi, (short)0, acc, false, false);
            acc = __builtin_amdgcn_wmma_f32_16x16x32_f16(
                      false, a_hi, false, b_lo, (short)0, acc, false, false);
            acc = __builtin_amdgcn_wmma_f32_16x16x32_f16(
                      false, a_lo, false, b_hi, (short)0, acc, false, false);
        }
    }

    // Epilogue: bias + ReLU; store per C/D layout (row = v + 8*hsel).
    const float bi = bias[oc];
    #pragma unroll
    for (int v = 0; v < 8; ++v) {
        const int mm = mtile * 16 + v + 8 * hsel;
        const int ni = mm / HW;
        const int r  = mm % HW;
        const int o  = ((ni * Cout + oc) * Hout + (r / Wout)) * Wout + (r % Wout);
        float y = acc[v] + bi;
        if (RELU) y = fmaxf(y, 0.0f);
        if constexpr (PACK_OUT) ((unsigned*)outv)[o] = pack_split(y);
        else                    ((float*)outv)[o] = y;
    }
}

// Sink head (16->36) and mask head (16->6 + softmax). One thread per batch row.
__global__ void heads_kernel(const float* __restrict__ latent,
                             const float* __restrict__ sink_w,
                             const float* __restrict__ sink_b,
                             const float* __restrict__ mask_w,
                             const float* __restrict__ mask_b,
                             float* __restrict__ log_alpha,
                             float* __restrict__ stopping,
                             int Bn)
{
    const int b = blockIdx.x * blockDim.x + threadIdx.x;
    if (b >= Bn) return;

    float lat[16];
    #pragma unroll
    for (int i = 0; i < 16; ++i) lat[i] = latent[b * 16 + i];

    #pragma unroll
    for (int j = 0; j < 36; ++j) {
        float s = sink_b[j];
        #pragma unroll
        for (int i = 0; i < 16; ++i) s += lat[i] * sink_w[j * 16 + i];
        log_alpha[b * 36 + j] = s;
    }

    float ml[6];
    float mx = -INFINITY;
    #pragma unroll
    for (int k = 0; k < 6; ++k) {
        float s = mask_b[k];
        #pragma unroll
        for (int i = 0; i < 16; ++i) s += lat[i] * mask_w[k * 16 + i];
        ml[k] = s;
        mx = fmaxf(mx, s);
    }
    float den = 0.0f;
    #pragma unroll
    for (int k = 0; k < 6; ++k) { ml[k] = expf(ml[k] - mx); den += ml[k]; }
    #pragma unroll
    for (int k = 0; k < 6; ++k) stopping[b * 6 + k] = ml[k] / den;
}

// Gumbel + 20-iteration Sinkhorn on a 6x6 in registers, then apply the
// transposed permutation to seq. One thread per (sample, batch) pair.
__global__ void sinkhorn_kernel(const float* __restrict__ log_alpha,
                                const float* __restrict__ gumbel,
                                const float* __restrict__ seq,
                                float* __restrict__ ordered,
                                int Bn, int NS)
{
    const int t = blockIdx.x * blockDim.x + threadIdx.x;
    if (t >= NS * Bn) return;
    const int b = t % Bn;

    float la[36];
    #pragma unroll
    for (int j = 0; j < 36; ++j)
        la[j] = log_alpha[b * 36 + j] + gumbel[t * 36 + j];  // noise=1, temp=1

    for (int it = 0; it < 20; ++it) {
        #pragma unroll
        for (int i = 0; i < 6; ++i) {              // row logsumexp (axis=2)
            float mx = la[i * 6 + 0];
            #pragma unroll
            for (int j = 1; j < 6; ++j) mx = fmaxf(mx, la[i * 6 + j]);
            float s = 0.0f;
            #pragma unroll
            for (int j = 0; j < 6; ++j) s += expf(la[i * 6 + j] - mx);
            const float lse = mx + logf(s);
            #pragma unroll
            for (int j = 0; j < 6; ++j) la[i * 6 + j] -= lse;
        }
        #pragma unroll
        for (int j = 0; j < 6; ++j) {              // col logsumexp (axis=1)
            float mx = la[0 * 6 + j];
            #pragma unroll
            for (int i = 1; i < 6; ++i) mx = fmaxf(mx, la[i * 6 + j]);
            float s = 0.0f;
            #pragma unroll
            for (int i = 0; i < 6; ++i) s += expf(la[i * 6 + j] - mx);
            const float lse = mx + logf(s);
            #pragma unroll
            for (int i = 0; i < 6; ++i) la[i * 6 + j] -= lse;
        }
    }

    #pragma unroll
    for (int i = 0; i < 6; ++i) {                  // ordered = sink^T @ seq
        float s = 0.0f;
        #pragma unroll
        for (int j = 0; j < 6; ++j)
            s += expf(la[j * 6 + i]) * seq[b * 6 + j];
        ordered[t * 6 + i] = s;
    }
}

extern "C" void kernel_launch(void* const* d_in, const int* in_sizes, int n_in,
                              void* d_out, int out_size, void* d_ws, size_t ws_size,
                              hipStream_t stream)
{
    (void)in_sizes; (void)n_in; (void)out_size; (void)ws_size;

    const float* seq    = (const float*)d_in[0];
    const float* im     = (const float*)d_in[1];
    const float* gumbel = (const float*)d_in[2];
    const float* w1 = (const float*)d_in[3];  const float* b1 = (const float*)d_in[4];
    const float* w2 = (const float*)d_in[5];  const float* b2 = (const float*)d_in[6];
    const float* w3 = (const float*)d_in[7];  const float* b3 = (const float*)d_in[8];
    const float* w4 = (const float*)d_in[9];  const float* b4 = (const float*)d_in[10];
    const float* enc_w  = (const float*)d_in[11]; const float* enc_b  = (const float*)d_in[12];
    const float* sink_w = (const float*)d_in[13]; const float* sink_b = (const float*)d_in[14];
    const float* mask_w = (const float*)d_in[15]; const float* mask_b = (const float*)d_in[16];

    float* out = (float*)d_out;            // [49152 ordered | 6144 stopping]
    unsigned* ws = (unsigned*)d_ws;        // 4-byte elements

    // Workspace layout (4B elems):
    //  R0 @ 0          (12,582,912): packed im; later conv2 out, conv4 out,
    //                                latent(f32)@+4,194,304, log_alpha@+5,242,880
    //  R1 @ 12,582,912 (29,491,200): conv1 out; later conv3 out
    //  RW @ 42,074,112 (~1.08M)    : packed weights w1,w2,w3,w4,enc
    unsigned* imP  = ws;
    unsigned* c1P  = ws + 12582912;
    unsigned* c2P  = ws;                   // im dead after conv1
    unsigned* c3P  = ws + 12582912;        // conv1 dead after conv2
    unsigned* c4P  = ws;                   // conv2 dead after conv3
    float*    lat  = (float*)(ws + 4194304);
    float*    lga  = (float*)(ws + 5242880);
    unsigned* w1P  = ws + 42074112;
    unsigned* w2P  = w1P + 2400;
    unsigned* w3P  = w2P + 51200;
    unsigned* w4P  = w3P + 204800;
    unsigned* encP = w4P + 819200;

    // ---- pack activations/weights into split-f16 uint32 form --------------
    pack_split_kernel<<<dim3(49152), dim3(256), 0, stream>>>(im, imP, 12582912);
    pack_split_kernel<<<dim3(10),    dim3(256), 0, stream>>>(w1, w1P, 2400);
    pack_split_kernel<<<dim3(200),   dim3(256), 0, stream>>>(w2, w2P, 51200);
    pack_split_kernel<<<dim3(800),   dim3(256), 0, stream>>>(w3, w3P, 204800);
    pack_split_kernel<<<dim3(3200),  dim3(256), 0, stream>>>(w4, w4P, 819200);
    pack_split_kernel<<<dim3(16),    dim3(256), 0, stream>>>(enc_w, encP, 4096);

    const dim3 blk(128);
    // conv1: M=921600 (57600 tiles /4), N=32 (2), K=75
    conv_wmma_kernel<3, 64, 64, 32, 30, 30, 5, 5, 2, true, true>
        <<<dim3(14400, 2), blk, 0, stream>>>(imP, w1P, b1, c1P);
    // conv2: M=173056 (10816 /4), N=64 (4), K=800
    conv_wmma_kernel<32, 30, 30, 64, 13, 13, 5, 5, 2, true, true>
        <<<dim3(2704, 4), blk, 0, stream>>>(c1P, w2P, b2, c2P);
    // conv3: M=25600 (1600 /4), N=128 (8), K=1600
    conv_wmma_kernel<64, 13, 13, 128, 5, 5, 5, 5, 2, true, true>
        <<<dim3(400, 8), blk, 0, stream>>>(c2P, w3P, b3, c3P);
    // conv4: M=1024 (64 /4), N=256 (16), K=3200
    conv_wmma_kernel<128, 5, 5, 256, 1, 1, 5, 5, 2, true, true>
        <<<dim3(16, 16), blk, 0, stream>>>(c3P, w4P, b4, c4P);
    // enc linear as 1x1, no ReLU, f32 out: M=1024 (64 /4), N=16 (1), K=256
    conv_wmma_kernel<256, 1, 1, 16, 1, 1, 1, 1, 1, false, false>
        <<<dim3(16, 1), blk, 0, stream>>>(c4P, encP, enc_b, lat);

    heads_kernel<<<dim3(4), dim3(256), 0, stream>>>(
        lat, sink_w, sink_b, mask_w, mask_b, lga, out + 49152, 1024);

    sinkhorn_kernel<<<dim3(32), dim3(256), 0, stream>>>(
        lga, gumbel, seq, out, 1024, 8);
}
